// MatmulAttention_22265110462921
// MI455X (gfx1250) — matrix operand order repaired and checked by reading, not compiled, not run on hardware
//
#include <hip/hip_runtime.h>

typedef __attribute__((ext_vector_type(16))) _Float16     v16h;
typedef __attribute__((ext_vector_type(8)))  float        v8f;
typedef __attribute__((ext_vector_type(4)))  float        f32x4;
typedef __attribute__((ext_vector_type(4)))  unsigned int u32x4;

namespace {
constexpr int S_SEQ = 2048;
constexpr int D_HEAD = 128;
constexpr int KT = 32;          // keys per tile
constexpr int NW = 8;           // waves per block
constexpr int QB = 16 * NW;     // 128 query rows per block

constexpr int KSTR = 136;       // LDS row stride (halves) for K tile   (multiple of 8)
constexpr int VSTR = 40;        // LDS row stride (halves) for V^T tile (multiple of 8)
constexpr int PSTR = 40;        // LDS row stride (halves) for P tile   (multiple of 8)

constexpr float LOG2E = 1.44269504088896340736f;
constexpr float MNEG  = -3.0e38f;   // finite "-inf" (fast-math safe)
}

union Frag {
  v16h     v;
  _Float16 h[16];
  u32x4    q[2];
};

#pragma float_control(precise, off)

// ---- single-instruction VOP2+DPP16 reduction steps (no canonicalize movs) ----
// masks {1,3,7,15} span GF(2)^4 -> full reduction over each 16-lane row.
__device__ __forceinline__ float vmax(float a, float b) {
  float d;
  asm("v_max_num_f32_e32 %0, %1, %2" : "=v"(d) : "v"(a), "v"(b));
  return d;
}
__device__ __forceinline__ float rmax16(float v) {
  float d;
  asm("v_max_num_f32_dpp %0, %1, %1 quad_perm:[1,0,3,2] row_mask:0xf bank_mask:0xf bound_ctrl:1"
      : "=v"(d) : "v"(v));
  asm("v_max_num_f32_dpp %0, %1, %1 quad_perm:[3,2,1,0] row_mask:0xf bank_mask:0xf bound_ctrl:1"
      : "=v"(d) : "v"(d));
  asm("v_max_num_f32_dpp %0, %1, %1 row_half_mirror row_mask:0xf bank_mask:0xf bound_ctrl:1"
      : "=v"(d) : "v"(d));
  asm("v_max_num_f32_dpp %0, %1, %1 row_mirror row_mask:0xf bank_mask:0xf bound_ctrl:1"
      : "=v"(d) : "v"(d));
  return d;
}
__device__ __forceinline__ float rsum16(float v) {
  float d;
  asm("v_add_f32_dpp %0, %1, %1 quad_perm:[1,0,3,2] row_mask:0xf bank_mask:0xf bound_ctrl:1"
      : "=v"(d) : "v"(v));
  asm("v_add_f32_dpp %0, %1, %1 quad_perm:[3,2,1,0] row_mask:0xf bank_mask:0xf bound_ctrl:1"
      : "=v"(d) : "v"(d));
  asm("v_add_f32_dpp %0, %1, %1 row_half_mirror row_mask:0xf bank_mask:0xf bound_ctrl:1"
      : "=v"(d) : "v"(d));
  asm("v_add_f32_dpp %0, %1, %1 row_mirror row_mask:0xf bank_mask:0xf bound_ctrl:1"
      : "=v"(d) : "v"(d));
  return d;
}

__global__ __launch_bounds__(256, 1)
void fa_fwd_f16wmma(const float* __restrict__ Q, const float* __restrict__ K,
                    const float* __restrict__ V, float* __restrict__ O)
{
  __shared__ __align__(16) _Float16 lk[2][KT * KSTR];      // K tile, row-major [key][d]
  __shared__ __align__(16) _Float16 lvt[2][D_HEAD * VSTR]; // V^T tile [d][perm-key]
  __shared__ __align__(16) _Float16 lp[NW][16 * PSTR];     // per-wave P tile [row][perm-key]

  const int tid  = threadIdx.x;
  const int lane = tid & 31;
  const int wv   = tid >> 5;
  const int n15  = lane & 15;
  const int hi   = lane >> 4;      // half-wave id
  const int kbA  = hi * 8;         // A-fragment K base (ISA 16-bit A layout)
  const int kbB  = hi * 8;         // B-fragment K base (ISA 16-bit B layout)

  const size_t base = (size_t)blockIdx.y * S_SEQ * D_HEAD;
  const float* qb  = Q + base;
  const float* kbp = K + base;
  const float* vbp = V + base;
  float*       ob  = O + base;

  const int    q0   = blockIdx.x * QB + wv * 16;
  const size_t qrow = (size_t)(q0 + n15);

  // staging geometry: 256 threads x 4 iters x float4 covers a 32x128 tile
  const int skey = tid >> 5;            // base key row this thread stages
  const int sd4  = (tid & 31) * 4;      // d offset this thread stages

  // ---- per-wave Q tile (16x128) as 4 f16 A-fragments, pre-scaled by log2(e) ----
  // (softmax computed in the log2 domain -> v_exp_f32 with no extra v_mul)
  Frag qa[4];
#pragma unroll
  for (int ks = 0; ks < 4; ++ks) {
    const float* p = qb + qrow * D_HEAD + ks * 32 + kbA;
    f32x4 f0 = ((const f32x4*)p)[0];
    f32x4 f1 = ((const f32x4*)p)[1];
    f32x4 f2 = ((const f32x4*)(p + 16))[0];
    f32x4 f3 = ((const f32x4*)(p + 16))[1];
    qa[ks].h[0]  = (_Float16)(f0.x * LOG2E);  qa[ks].h[1]  = (_Float16)(f0.y * LOG2E);
    qa[ks].h[2]  = (_Float16)(f0.z * LOG2E);  qa[ks].h[3]  = (_Float16)(f0.w * LOG2E);
    qa[ks].h[4]  = (_Float16)(f1.x * LOG2E);  qa[ks].h[5]  = (_Float16)(f1.y * LOG2E);
    qa[ks].h[6]  = (_Float16)(f1.z * LOG2E);  qa[ks].h[7]  = (_Float16)(f1.w * LOG2E);
    qa[ks].h[8]  = (_Float16)(f2.x * LOG2E);  qa[ks].h[9]  = (_Float16)(f2.y * LOG2E);
    qa[ks].h[10] = (_Float16)(f2.z * LOG2E);  qa[ks].h[11] = (_Float16)(f2.w * LOG2E);
    qa[ks].h[12] = (_Float16)(f3.x * LOG2E);  qa[ks].h[13] = (_Float16)(f3.y * LOG2E);
    qa[ks].h[14] = (_Float16)(f3.z * LOG2E);  qa[ks].h[15] = (_Float16)(f3.w * LOG2E);
  }

  v8f o[8] = {};                 // 16x128 fp32 output accumulator (8 d-chunks)
  float mrow[8], lrow[8];        // running row max (log2 units) / row sum
#pragma unroll
  for (int r = 0; r < 8; ++r) { mrow[r] = MNEG; lrow[r] = 0.f; }

  // ---- prefetch tile 0 (fp32) into registers ----
  f32x4 pk[4], pv[4];
#pragma unroll
  for (int it = 0; it < 4; ++it) {
    const size_t g = (size_t)(skey + it * 8) * D_HEAD + sd4;
    pk[it] = *(const f32x4*)(kbp + g);
    pv[it] = *(const f32x4*)(vbp + g);
  }

  int buf = 0;
  for (int kt0 = 0; kt0 < S_SEQ; kt0 += KT) {
    // ---- commit prefetched tile: fp32 -> f16, K row-major + V^T (perm cols) ----
#pragma unroll
    for (int it = 0; it < 4; ++it) {
      const int key  = skey + it * 8;
      const int vcol = ((key & 15) << 1) | (key >> 4);   // interleaved key order
      _Float16* kd = &lk[buf][key * KSTR + sd4];
      kd[0] = (_Float16)pk[it].x; kd[1] = (_Float16)pk[it].y;
      kd[2] = (_Float16)pk[it].z; kd[3] = (_Float16)pk[it].w;
      lvt[buf][(sd4 + 0) * VSTR + vcol] = (_Float16)pv[it].x;
      lvt[buf][(sd4 + 1) * VSTR + vcol] = (_Float16)pv[it].y;
      lvt[buf][(sd4 + 2) * VSTR + vcol] = (_Float16)pv[it].z;
      lvt[buf][(sd4 + 3) * VSTR + vcol] = (_Float16)pv[it].w;
    }
    __syncthreads();

    // ---- issue next tile's global loads; retire during compute below ----
    if (kt0 + KT < S_SEQ) {
#pragma unroll
      for (int it = 0; it < 4; ++it) {
        const size_t g = (size_t)(kt0 + KT + skey + it * 8) * D_HEAD + sd4;
        pk[it] = *(const f32x4*)(kbp + g);
        pv[it] = *(const f32x4*)(vbp + g);
      }
    }

    // ---- S = Q * K^T : two 16x16 chunks over the 32-key tile ----
    v8f s0 = {}, s1 = {};
#pragma unroll
    for (int ks = 0; ks < 4; ++ks) {
      Frag b0, b1;
      const _Float16* r0 = &lk[buf][n15 * KSTR + ks * 32 + kbB];
      const _Float16* r1 = &lk[buf][(16 + n15) * KSTR + ks * 32 + kbB];
      b0.q[0] = *(const u32x4*)r0;  b0.q[1] = *(const u32x4*)(r0 + 16);
      b1.q[0] = *(const u32x4*)r1;  b1.q[1] = *(const u32x4*)(r1 + 16);
      s0 = __builtin_amdgcn_wmma_f32_16x16x32_f16(false, qa[ks].v, false, b0.v,
                                                  (short)0, s0, false, false);
      s1 = __builtin_amdgcn_wmma_f32_16x16x32_f16(false, qa[ks].v, false, b1.v,
                                                  (short)0, s1, false, false);
    }

    // ---- online softmax in log2 domain (row = r + 8*hi) ----
    float alpha[8];
#pragma unroll
    for (int r = 0; r < 8; ++r) {
      float mx = rmax16(vmax(s0[r], s1[r]));
      float mn = vmax(mrow[r], mx);
      alpha[r] = __builtin_amdgcn_exp2f(mrow[r] - mn);
      mrow[r] = mn;
    }
#pragma unroll
    for (int r = 0; r < 8; ++r) {
      float p0 = __builtin_amdgcn_exp2f(s0[r] - mrow[r]);
      float p1 = __builtin_amdgcn_exp2f(s1[r] - mrow[r]);
      lrow[r] = lrow[r] * alpha[r] + rsum16(p0 + p1);
      // pack both probs (perm-adjacent keys n15 and 16+n15) into one b32 store
      union { _Float16 h[2]; unsigned int u; } pp;
      pp.h[0] = (_Float16)p0;
      pp.h[1] = (_Float16)p1;
      const int row = r + 8 * hi;
      *(unsigned int*)&lp[wv][row * PSTR + 2 * n15] = pp.u;
    }

    // ---- reload P in A-fragment layout (16x32 over permuted key axis) ----
    Frag pa;
    const _Float16* pr = &lp[wv][n15 * PSTR + kbA];
    pa.q[0] = *(const u32x4*)pr;
    pa.q[1] = *(const u32x4*)(pr + 16);

    // ---- O = diag(alpha)*O + P*V  (8 d-chunks of 16; V^T cols share the perm) ----
#pragma unroll
    for (int c = 0; c < 8; ++c) {
      Frag bv;
      const _Float16* vr = &lvt[buf][(c * 16 + n15) * VSTR + kbB];
      bv.q[0] = *(const u32x4*)vr;
      bv.q[1] = *(const u32x4*)(vr + 16);
      v8f cc;
#pragma unroll
      for (int r = 0; r < 8; ++r) cc[r] = o[c][r] * alpha[r];
      o[c] = __builtin_amdgcn_wmma_f32_16x16x32_f16(false, pa.v, false, bv.v,
                                                    (short)0, cc, false, false);
    }
    buf ^= 1;
  }

  // ---- epilogue: normalize by row sums (1 rcp per row) and store fp32 ----
  float rinv[8];
#pragma unroll
  for (int r = 0; r < 8; ++r) rinv[r] = 1.0f / lrow[r];
#pragma unroll
  for (int c = 0; c < 8; ++c) {
#pragma unroll
    for (int r = 0; r < 8; ++r) {
      const size_t row = (size_t)(q0 + r + 8 * hi);
      ob[row * D_HEAD + c * 16 + n15] = o[c][r] * rinv[r];
    }
  }
}

extern "C" void kernel_launch(void* const* d_in, const int* in_sizes, int n_in,
                              void* d_out, int out_size, void* d_ws, size_t ws_size,
                              hipStream_t stream) {
  const float* q = (const float*)d_in[0];
  const float* k = (const float*)d_in[1];
  const float* v = (const float*)d_in[2];
  float* out = (float*)d_out;
  const int bh = in_sizes[0] / (S_SEQ * D_HEAD);   // B*H = 64
  dim3 grid(S_SEQ / QB, bh);
  fa_fwd_f16wmma<<<grid, 256, 0, stream>>>(q, k, v, out);
}
